// SpMiddleFHD_65738769432596
// MI455X (gfx1250) — matrix operand order repaired and checked by reading, not compiled, not run on hardware
//
#include <hip/hip_runtime.h>

#ifndef __has_builtin
#define __has_builtin(x) 0
#endif

// ---------------------------------------------------------------------------
// CDNA5 (gfx1250) implicit-GEMM sparse-middle-encoder.
// NDHWC bf16 activations; weights pre-packed to WMMA A-fragment stream and
// DMA'd to LDS per block via the Tensor Data Mover (TDM).
// Wave computes an MT(cout-blocks) x NT(W-tiles) grid of 16x16 tiles.
// Wave-uniform interior/border split: interior blocks run a predicate-free
// inner loop (pure load/ds/wmma).
// ---------------------------------------------------------------------------

typedef __attribute__((ext_vector_type(16))) __bf16          v16bf;
typedef __attribute__((ext_vector_type(16))) unsigned short  v16u;
typedef __attribute__((ext_vector_type(8)))  unsigned short  v8u;
typedef __attribute__((ext_vector_type(8)))  float           v8f;
typedef __attribute__((ext_vector_type(4)))  unsigned int    v4u;
typedef __attribute__((ext_vector_type(8)))  int             v8i;
typedef __attribute__((ext_vector_type(4)))  int             v4i;

template <bool B> struct BoolC { static constexpr bool value = B; };

__device__ __forceinline__ unsigned short f2bf(float f) {
    unsigned int u = __builtin_bit_cast(unsigned int, f);
    return (unsigned short)(u >> 16);
}

// One TDM 1D transfer: global -> LDS, nbytes (8B-aligned), then wait.
__device__ __forceinline__ void stage_weights(const unsigned short* gsrc,
                                              unsigned short* ldst,
                                              unsigned nbytes, int lane) {
#if __has_builtin(__builtin_amdgcn_tensor_load_to_lds)
    const unsigned long long ga = (unsigned long long)(const void*)gsrc;
    const unsigned lo = (unsigned)(unsigned long long)(void*)ldst; // LDS offset
    const unsigned el = nbytes >> 3;                               // 8B elements
    v4u g0;
    g0[0] = 1u;                                    // count=1, user mode
    g0[1] = lo;                                    // lds_addr
    g0[2] = (unsigned)ga;                          // global_addr[31:0]
    g0[3] = ((unsigned)(ga >> 32) & 0x1ffffffu) | (2u << 30); // [56:32]|type=2
    v8i g1;
    g1[0] = (int)(3u << 16);                       // data_size = 8 bytes
    g1[1] = (int)((el & 0xffffu) << 16);           // tensor_dim0[15:0]
    g1[2] = (int)(((el >> 16) & 0xffffu) | (1u << 16)); // dim0[31:16], dim1=1
    g1[3] = (int)((el & 0xffffu) << 16);           // tile_dim0 = el
    g1[4] = 0;                                     // tile_dim1/2 unused
    g1[5] = (int)el;                               // tensor_dim0_stride[31:0]
    g1[6] = 0;
    g1[7] = 0;
    v4i gz = {};
#if __clang_major__ >= 23
    v8i gz8 = {};
    __builtin_amdgcn_tensor_load_to_lds(g0, g1, gz, gz, gz8, 0);
#else
    __builtin_amdgcn_tensor_load_to_lds(g0, g1, gz, gz, 0);
#endif
    __builtin_amdgcn_s_wait_tensorcnt(0);
    (void)lane;
#else
    const v16u* s = (const v16u*)gsrc;
    v16u*       d = (v16u*)ldst;
    const unsigned nv = nbytes >> 5;
    for (unsigned i = lane; i < nv; i += 32) d[i] = s[i];
#endif
}

// ---------------------------------------------------------------------------
__global__ void scatter_kernel(const float* __restrict__ feats,
                               const int*   __restrict__ coors,
                               float* __restrict__ x0,
                               float* __restrict__ mask0,
                               int nv, int D, int H, int W) {
    int t = blockIdx.x * blockDim.x + threadIdx.x;
    if (t >= nv * 16) return;
    int vi = t >> 4;
    int c  = t & 15;
    int b = coors[4 * vi + 0];
    int z = coors[4 * vi + 1];
    int y = coors[4 * vi + 2];
    int x = coors[4 * vi + 3];
    size_t sp = (((size_t)b * D + z) * H + y) * W + x;
    atomicAdd(&x0[sp * 16 + c], feats[t]);
    if (c == 0) mask0[sp] = 1.0f;
}

__global__ void f32_to_bf16_kernel(const float* __restrict__ in,
                                   unsigned short* __restrict__ out,
                                   long long n) {
    long long i = (long long)blockIdx.x * blockDim.x + threadIdx.x;
    if (i < n) out[i] = f2bf(in[i]);
}

// ---------------------------------------------------------------------------
__global__ void mask_pool_kernel(const float* __restrict__ min_,
                                 float* __restrict__ mout,
                                 int N, int Din, int Hin, int Win,
                                 int Dout, int Hout, int Wout,
                                 int kd, int kh, int kw,
                                 int sd, int sh, int sw,
                                 int pd, int ph, int pw) {
    long long idx   = (long long)blockIdx.x * blockDim.x + threadIdx.x;
    long long total = (long long)N * Dout * Hout * Wout;
    if (idx >= total) return;
    int wo = idx % Wout;
    int ho = (idx / Wout) % Hout;
    int dO = (idx / ((long long)Wout * Hout)) % Dout;
    int n  = idx / ((long long)Wout * Hout * Dout);
    float any = 0.0f;
    for (int kz = 0; kz < kd; ++kz) {
        int zi = dO * sd - pd + kz;
        if (zi < 0 || zi >= Din) continue;
        for (int ky = 0; ky < kh; ++ky) {
            int yi = ho * sh - ph + ky;
            if (yi < 0 || yi >= Hin) continue;
            for (int kx = 0; kx < kw; ++kx) {
                int xi = wo * sw - pw + kx;
                if (xi < 0 || xi >= Win) continue;
                if (min_[(((size_t)n * Din + zi) * Hin + yi) * (size_t)Win + xi] > 0.0f)
                    any = 1.0f;
            }
        }
    }
    mout[idx] = any;
}

// ---------------------------------------------------------------------------
// Pack layer weights (f32 OIDHW) -> WMMA A-fragment stream (bf16).
// ---------------------------------------------------------------------------
template <int CIN, int KVOL>
__global__ void pack_weights_kernel(const float* __restrict__ wgt,
                                    unsigned short* __restrict__ wpk,
                                    int Cout) {
    constexpr int NCHUNK = (CIN == 16) ? (KVOL + 1) / 2
                         : (CIN == 32) ? KVOL : 2 * KVOL;
    int i = blockIdx.x * blockDim.x + threadIdx.x;
    int total = (Cout >> 4) * NCHUNK * 512;
    if (i >= total) return;
    const int e    = i & 15;
    const int lt   = (i >> 4) & 31;
    const int chcb = i >> 9;
    const int ch   = chcb % NCHUNK;
    const int cb   = chcb / NCHUNK;
    const int hit  = lt >> 4;
    const int colt = lt & 15;
    const int vg   = e >> 1;
    const int part = e & 1;
    const int K = 16 * hit + 2 * vg + part;
    int tap, ci;
    if (CIN == 16)      { tap = 2 * ch + (K >> 4); ci = K & 15; }
    else if (CIN == 32) { tap = ch;                ci = K; }
    else                { tap = ch >> 1;           ci = ((ch & 1) << 5) + K; }
    unsigned short val = 0;
    if (tap < KVOL)
        val = f2bf(wgt[((size_t)(cb * 16 + colt) * CIN + ci) * KVOL + tap]);
    wpk[i] = val;
}

// ---------------------------------------------------------------------------
// Implicit-GEMM conv3d + BN + ReLU + mask, MT x NT register tiling.
// ---------------------------------------------------------------------------
template <int MT, int NT, int CIN, int COUT, int KD, int KH, int KW,
          int SD, int SH, int SW, int PD, int PH, int PW,
          int DD, int DH, int DW,
          int DIN, int HIN, int WIN, int DOUT, int HOUT, int WOUT,
          bool FINAL>
__global__ __launch_bounds__(32)
void conv_bnrelu_wmma_kernel(const unsigned short* __restrict__ xin,
                             const unsigned short* __restrict__ wpk,
                             const float* __restrict__ gg,
                             const float* __restrict__ bb,
                             const float* __restrict__ mm,
                             const float* __restrict__ vv,
                             const float* __restrict__ mask,
                             unsigned short* __restrict__ xout,
                             float* __restrict__ outf) {
    constexpr int KVOL   = KD * KH * KW;
    constexpr int NCHUNK = (CIN == 16) ? (KVOL + 1) / 2
                         : (CIN == 32) ? KVOL : 2 * KVOL;
    constexpr int COB    = COUT / 16;
    constexpr int NMB    = COB / MT;

    const int lane = threadIdx.x;
    const int hi   = lane >> 4;
    const int col  = lane & 15;

    const int w0  = blockIdx.x * (16 * NT);
    const int dO  = blockIdx.y / HOUT;
    const int hO  = blockIdx.y % HOUT;
    const int n   = blockIdx.z / NMB;
    const int cb0 = (blockIdx.z % NMB) * MT;

    // ---- TDM: DMA MT contiguous fragment slabs into LDS ----
    extern __shared__ unsigned short smem[];
    stage_weights(wpk + (size_t)cb0 * NCHUNK * 512, smem,
                  (unsigned)(MT * NCHUNK * 1024), lane);

    // ---- Predicate tables (used only on the border path) ----
    bool zok[KD], yok[KH], xok[NT][KW], wok[NT];
#pragma unroll
    for (int kz = 0; kz < KD; ++kz) {
        int zi = dO * SD - PD + kz * DD;
        zok[kz] = (zi >= 0) && (zi < DIN);
    }
#pragma unroll
    for (int ky = 0; ky < KH; ++ky) {
        int yi = hO * SH - PH + ky * DH;
        yok[ky] = (yi >= 0) && (yi < HIN);
    }
    const unsigned short* bptr[NT];
#pragma unroll
    for (int j = 0; j < NT; ++j) {
        const int wo = w0 + j * 16 + col;
        wok[j] = wo < WOUT;
#pragma unroll
        for (int kx = 0; kx < KW; ++kx) {
            int xi = wo * SW - PW + kx * DW;
            xok[j][kx] = (xi >= 0) && (xi < WIN) && wok[j];
        }
        int off0 = (((n * DIN + (dO * SD - PD)) * HIN + (hO * SH - PH)) * WIN +
                    (wo * SW - PW)) * CIN;
        if (CIN >= 32) off0 += 16 * hi;
        bptr[j] = xin + (ptrdiff_t)off0;
    }

    // ---- Wave-uniform interior test (w0/dO/hO only -> scalar branch) ----
    const int wLast = w0 + NT * 16 - 1;
    const bool interior =
        (dO * SD - PD >= 0) && (dO * SD - PD + (KD - 1) * DD < DIN) &&
        (hO * SH - PH >= 0) && (hO * SH - PH + (KH - 1) * DH < HIN) &&
        (w0 * SW - PW >= 0) && (wLast * SW - PW + (KW - 1) * DW < WIN) &&
        (wLast < WOUT);

    v8f acc[MT][NT];
#pragma unroll
    for (int mt = 0; mt < MT; ++mt)
#pragma unroll
        for (int j = 0; j < NT; ++j) acc[mt][j] = (v8f){};

    auto chunk_loop = [&](auto gen) {
        constexpr bool GENERAL = decltype(gen)::value;
#pragma unroll
        for (int ch = 0; ch < NCHUNK; ++ch) {
            int boff;
            int kz0 = 0, ky0 = 0, kx0 = 0, kz1 = 0, ky1 = 0, kx1 = 0;
            bool oddTail = false;
            if (CIN == 16) {
                const int t0  = 2 * ch;
                const int t1  = 2 * ch + 1;
                oddTail = (t1 >= KVOL);
                const int t1c = oddTail ? 0 : t1;
                kz0 = t0 / (KH * KW); ky0 = (t0 / KW) % KH; kx0 = t0 % KW;
                kz1 = t1c / (KH * KW); ky1 = (t1c / KW) % KH; kx1 = t1c % KW;
                const int b0 = ((kz0 * DD * HIN + ky0 * DH) * WIN + kx0 * DW) * CIN;
                const int b1 = ((kz1 * DD * HIN + ky1 * DH) * WIN + kx1 * DW) * CIN;
                boff = hi ? b1 : b0;
            } else {
                const int tap = (CIN == 32) ? ch : (ch >> 1);
                kz0 = tap / (KH * KW); ky0 = (tap / KW) % KH; kx0 = tap % KW;
                boff = ((kz0 * DD * HIN + ky0 * DH) * WIN + kx0 * DW) * CIN +
                       ((CIN == 64) ? ((ch & 1) << 5) : 0);
            }

            v16u bu[NT];
            if (!GENERAL && !oddTail) {
                // interior fast path: unconditional loads
#pragma unroll
                for (int j = 0; j < NT; ++j)
                    bu[j] = *(const v16u*)(bptr[j] + boff);
            } else {
                bool v[NT];
#pragma unroll
                for (int j = 0; j < NT; ++j) {
                    if (GENERAL) {
                        if (CIN == 16) {
                            const bool v0 = zok[kz0] && yok[ky0] && xok[j][kx0];
                            const bool v1 = !oddTail && zok[kz1] && yok[ky1] &&
                                            xok[j][kx1];
                            v[j] = hi ? v1 : v0;
                        } else {
                            v[j] = zok[kz0] && yok[ky0] && xok[j][kx0];
                        }
                    } else {
                        v[j] = !hi;   // interior, CIN16 odd tail: upper half void
                    }
                }
#pragma unroll
                for (int j = 0; j < NT; ++j) {
                    v16u z = {};
                    bu[j] = z;
                    if (v[j]) bu[j] = *(const v16u*)(bptr[j] + boff);
                }
            }

            v16u au[MT];
#pragma unroll
            for (int mt = 0; mt < MT; ++mt)
                au[mt] = *(const v16u*)(smem + ((mt * NCHUNK + ch) << 9) +
                                        (lane << 4));

#pragma unroll
            for (int mt = 0; mt < MT; ++mt)
#pragma unroll
                for (int j = 0; j < NT; ++j)
                    acc[mt][j] = __builtin_amdgcn_wmma_f32_16x16x32_bf16(
                        false, __builtin_bit_cast(v16bf, au[mt]),
                        false, __builtin_bit_cast(v16bf, bu[j]),
                        (short)0, acc[mt][j], false, false);
        }
    };

    if (interior) chunk_loop(BoolC<false>{});
    else          chunk_loop(BoolC<true>{});

    // ---- Epilogue: BN + ReLU + mask ----
    float  mval[NT];
    size_t spo[NT];
#pragma unroll
    for (int j = 0; j < NT; ++j) {
        const int wo = w0 + j * 16 + col;
        spo[j] = (((size_t)n * DOUT + dO) * HOUT + hO) * (size_t)WOUT + wo;
        mval[j] = 0.0f;
        if (wok[j]) mval[j] = mask[spo[j]];
    }

#pragma unroll
    for (int mt = 0; mt < MT; ++mt) {
        const int cbase8 = (cb0 + mt) * 16 + 8 * hi;
        const v8f gv  = *(const v8f*)(gg + cbase8);
        const v8f bv  = *(const v8f*)(bb + cbase8);
        const v8f mv  = *(const v8f*)(mm + cbase8);
        const v8f vvv = *(const v8f*)(vv + cbase8);
        float s[8], bi[8];
#pragma unroll
        for (int vg = 0; vg < 8; ++vg) {
            s[vg]  = gv[vg] * rsqrtf(vvv[vg] + 1e-3f);
            bi[vg] = bv[vg] - mv[vg] * s[vg];
        }
#pragma unroll
        for (int j = 0; j < NT; ++j) {
            float r[8];
#pragma unroll
            for (int vg = 0; vg < 8; ++vg)
                r[vg] = fmaxf(acc[mt][j][vg] * s[vg] + bi[vg], 0.0f) * mval[j];

            if constexpr (!FINAL) {
                if (wok[j]) {
                    v8u pk;
#pragma unroll
                    for (int vg = 0; vg < 8; ++vg) pk[vg] = f2bf(r[vg]);
                    *(v8u*)(xout + spo[j] * COUT + cbase8) = pk;
                }
            } else {
                constexpr size_t SPOUT = (size_t)DOUT * HOUT * WOUT;
                const int wo = w0 + j * 16 + col;
                if (wok[j]) {
#pragma unroll
                    for (int vg = 0; vg < 8; ++vg) {
                        const int co = cbase8 + vg;
                        outf[((size_t)n * COUT + co) * SPOUT +
                             ((size_t)dO * HOUT + hO) * (size_t)WOUT + wo] = r[vg];
                    }
                }
            }
        }
    }
}

// ---------------------------------------------------------------------------
// Host side
// ---------------------------------------------------------------------------
template <int MT, int NT, int CIN, int COUT, int KD, int KH, int KW,
          int SD, int SH, int SW, int PD, int PH, int PW,
          int DD, int DH, int DW,
          int DIN, int HIN, int WIN, int DOUT, int HOUT, int WOUT,
          bool FINAL, bool SPCONV>
static inline void run_layer(const unsigned short* xin, unsigned short* xout,
                             float* outf,
                             const float* w, const float* g, const float* b,
                             const float* m, const float* v,
                             const float* mask_in, float* mask_out,
                             unsigned short* wpack, hipStream_t stream) {
    constexpr int KVOL   = KD * KH * KW;
    constexpr int NCHUNK = (CIN == 16) ? (KVOL + 1) / 2
                         : (CIN == 32) ? KVOL : 2 * KVOL;
    constexpr int COB    = COUT / 16;

    const int total = COB * NCHUNK * 512;
    pack_weights_kernel<CIN, KVOL><<<(total + 255) / 256, 256, 0, stream>>>(
        w, wpack, COUT);

    if (SPCONV) {
        long long mt = 2LL * DOUT * HOUT * WOUT;
        mask_pool_kernel<<<(int)((mt + 255) / 256), 256, 0, stream>>>(
            mask_in, mask_out, 2, DIN, HIN, WIN, DOUT, HOUT, WOUT,
            KD, KH, KW, SD, SH, SW, PD, PH, PW);
    }

    const int wtiles = (WOUT + 15) / 16;
    dim3 grid((wtiles + NT - 1) / NT, DOUT * HOUT, 2 * (COB / MT));
    const size_t shbytes = (size_t)MT * NCHUNK * 1024;
    conv_bnrelu_wmma_kernel<MT, NT, CIN, COUT, KD, KH, KW, SD, SH, SW,
                            PD, PH, PW, DD, DH, DW,
                            DIN, HIN, WIN, DOUT, HOUT, WOUT, FINAL>
        <<<grid, 32, shbytes, stream>>>(xin, wpack, g, b, m, v,
                                        SPCONV ? mask_out : mask_in, xout, outf);
}

extern "C" void kernel_launch(void* const* d_in, const int* in_sizes, int n_in,
                              void* d_out, int out_size, void* d_ws, size_t ws_size,
                              hipStream_t stream) {
    const float* feats = (const float*)d_in[0];
    const int*   coors = (const int*)d_in[1];
    const int NV = in_sizes[0] / 16;
    const int D0 = 41, H0 = 256, W0 = 224;

    const size_t BUF  = (size_t)2 * 16 * D0 * H0 * W0;
    const size_t MBUF = (size_t)2 * D0 * H0 * W0;

    float*          x0f   = (float*)d_ws;
    unsigned short* bufA  = (unsigned short*)(x0f + BUF);
    unsigned short* bufB  = (unsigned short*)x0f;          // aliases x0f
    float*          maskA = (float*)(bufA + BUF);
    float*          maskB = maskA + MBUF;
    unsigned short* wpack = (unsigned short*)(maskB + MBUF);

    hipMemsetAsync(x0f,   0, BUF  * sizeof(float), stream);
    hipMemsetAsync(maskA, 0, MBUF * sizeof(float), stream);

    scatter_kernel<<<(NV * 16 + 255) / 256, 256, 0, stream>>>(
        feats, coors, x0f, maskA, NV, D0, H0, W0);
    f32_to_bf16_kernel<<<(int)((BUF + 255) / 256), 256, 0, stream>>>(
        x0f, bufA, (long long)BUF);

    unsigned short* A  = bufA;
    unsigned short* B  = bufB;
    float*          mA = maskA;
    float*          mB = maskB;
    float*          OF = (float*)d_out;

#define PP(l) (const float*)d_in[3 + 5 * (l) + 0], (const float*)d_in[3 + 5 * (l) + 1], \
              (const float*)d_in[3 + 5 * (l) + 2], (const float*)d_in[3 + 5 * (l) + 3], \
              (const float*)d_in[3 + 5 * (l) + 4]
#define SWAPB { unsigned short* t = A; A = B; B = t; }
#define SWAPM { float* t = mA; mA = mB; mB = t; }

    run_layer<1,2, 16,16,3,3,3, 1,1,1, 1,1,1, 1,1,1, 41,256,224, 41,256,224, false,false>
        (A, B, nullptr, PP(0), mA, mB, wpack, stream); SWAPB;
    run_layer<1,2, 16,16,3,3,3, 1,1,1, 2,2,2, 2,2,2, 41,256,224, 41,256,224, false,false>
        (A, B, nullptr, PP(1), mA, mB, wpack, stream); SWAPB;
    run_layer<2,2, 16,32,3,3,3, 2,2,2, 1,1,1, 1,1,1, 41,256,224, 21,128,112, false,true>
        (A, B, nullptr, PP(2), mA, mB, wpack, stream); SWAPB; SWAPM;
    run_layer<2,2, 32,32,3,3,3, 1,1,1, 1,1,1, 1,1,1, 21,128,112, 21,128,112, false,false>
        (A, B, nullptr, PP(3), mA, mB, wpack, stream); SWAPB;
    run_layer<2,2, 32,32,3,3,3, 1,1,1, 2,2,2, 2,2,2, 21,128,112, 21,128,112, false,false>
        (A, B, nullptr, PP(4), mA, mB, wpack, stream); SWAPB;
    run_layer<2,2, 32,64,3,3,3, 2,2,2, 1,1,1, 1,1,1, 21,128,112, 11,64,56, false,true>
        (A, B, nullptr, PP(5), mA, mB, wpack, stream); SWAPB; SWAPM;
    run_layer<2,2, 64,64,3,3,3, 1,1,1, 1,1,1, 1,1,1, 11,64,56, 11,64,56, false,false>
        (A, B, nullptr, PP(6), mA, mB, wpack, stream); SWAPB;
    run_layer<2,2, 64,64,3,3,3, 1,1,1, 2,2,2, 2,2,2, 11,64,56, 11,64,56, false,false>
        (A, B, nullptr, PP(7), mA, mB, wpack, stream); SWAPB;
    run_layer<2,2, 64,64,3,3,3, 1,1,1, 3,3,3, 3,3,3, 11,64,56, 11,64,56, false,false>
        (A, B, nullptr, PP(8), mA, mB, wpack, stream); SWAPB;
    run_layer<2,2, 64,64,3,3,3, 2,2,2, 0,1,1, 1,1,1, 11,64,56, 5,32,28, false,true>
        (A, B, nullptr, PP(9), mA, mB, wpack, stream); SWAPB; SWAPM;
    run_layer<2,2, 64,64,3,3,3, 1,1,1, 1,1,1, 1,1,1, 5,32,28, 5,32,28, false,false>
        (A, B, nullptr, PP(10), mA, mB, wpack, stream); SWAPB;
    run_layer<2,2, 64,64,3,3,3, 1,1,1, 2,2,2, 2,2,2, 5,32,28, 5,32,28, false,false>
        (A, B, nullptr, PP(11), mA, mB, wpack, stream); SWAPB;
    run_layer<2,2, 64,64,3,3,3, 1,1,1, 3,3,3, 3,3,3, 5,32,28, 5,32,28, false,false>
        (A, B, nullptr, PP(12), mA, mB, wpack, stream); SWAPB;
    run_layer<2,2, 64,64,3,1,1, 2,1,1, 0,0,0, 1,1,1, 5,32,28, 2,32,28, true,true>
        (A, nullptr, OF, PP(13), mA, mB, wpack, stream);

#undef PP
#undef SWAPB
#undef SWAPM
    (void)n_in; (void)out_size; (void)ws_size;
}
